// PerspectiveNet768x2_59064390255175
// MI455X (gfx1250) — compile-verified, matrix operand order checked by reading, not run on hardware
//
#include <hip/hip_runtime.h>
#include <hip/hip_bf16.h>

// PerspectiveNet768x2 (NNUE-style) for MI455X / gfx1250.
//
// Design (see analysis): L2-bandwidth-bound sparse gather-accumulate.
//  - 1 wave32 == 1 batch row; full 1024-f32 accumulators per side in VGPRs.
//  - global_load_b128 gathers (weights are L2-resident: 31.5 MB << 192 MB L2).
//  - global_prefetch_b8 on next feature row.
//  - Output layer (the only true matrix op) via v_wmma_f32_16x16x32_f16:
//    row-sum of prod[16 x 2048] as A x ones(32x16), f32 accumulation in C.

typedef __attribute__((ext_vector_type(16))) _Float16 v16h;
typedef __attribute__((ext_vector_type(8)))  _Float16 v8h;
typedef __attribute__((ext_vector_type(4)))  _Float16 v4h;
typedef __attribute__((ext_vector_type(8)))  float    v8f;

#define HIDDEN   1024
#define MAXA     32
#define BPB      16        // batch rows per block (== WMMA M)
#define RSTRIDE  2056      // f16 LDS row stride: 2048 + 8 pad -> 16B aligned,
                           // 4-bank rotation per row => conflict-free b128 reads

__global__ __launch_bounds__(512) void nnue_ft_kernel(
    const int*  __restrict__ fw,            // [B,32] white feature ids (-1 pad)
    const int*  __restrict__ fb,            // [B,32] black feature ids (-1 pad)
    const unsigned char* __restrict__ stm,  // [B] jnp.bool_ (1 byte)
    const float* __restrict__ Ww,           // [3840,1024]
    const float* __restrict__ bw,           // [1024]
    const float* __restrict__ Wb,           // [3840,1024]
    const float* __restrict__ bb,           // [1024]
    const float* __restrict__ ow,           // [2048]
    const float* __restrict__ ob,           // [1]
    float* __restrict__ out)                // [B]
{
    __shared__ __align__(16) _Float16 sprod[BPB][RSTRIDE];
    __shared__ float sOut[BPB];

    const int tid  = threadIdx.x;
    const int lane = tid & 31;
    const int wave = tid >> 5;                 // 0..15
    const int b    = blockIdx.x * BPB + wave;  // batch row owned by this wave

    if (tid < BPB) sOut[tid] = 0.0f;

    // ---- preload this row's feature indices: one per lane -----------------
    const int myfw = fw[b * MAXA + lane];
    const int myfb = fb[b * MAXA + lane];

    // ---- f32 accumulators, initialized with the FT biases -----------------
    // lane l, chunk c covers hidden cols [c*128 + l*4, +4)
    const float4* bw4 = (const float4*)bw;
    const float4* bb4 = (const float4*)bb;
    float4 accw[8], accb[8];
#pragma unroll
    for (int c = 0; c < 8; ++c) {
        accw[c] = bw4[c * 32 + lane];
        accb[c] = bb4[c * 32 + lane];
    }

    // ---- sparse gather-accumulate (dominant cost: ~4.3 GB from L2) --------
    for (int a = 0; a < MAXA; ++a) {
        const int iw = __shfl(myfw, a);   // uniform across the wave
        if (iw < 0) break;                // trailing -1 padding: uniform exit
        const int ib = __shfl(myfb, a);

        const float4* rw = (const float4*)(Ww + (size_t)iw * HIDDEN);
        const float4* rb = (const float4*)(Wb + (size_t)ib * HIDDEN);

        // prefetch next feature's rows (global_prefetch_b8); lane*128B strides
        // cover the full 4KB row across the wave
        if (a + 1 < MAXA) {
            const int iwn = __shfl(myfw, a + 1);
            if (iwn >= 0) {
                const int ibn = __shfl(myfb, a + 1);
                __builtin_prefetch(Ww + (size_t)iwn * HIDDEN + lane * 32, 0, 0);
                __builtin_prefetch(Wb + (size_t)ibn * HIDDEN + lane * 32, 0, 0);
            }
        }

#pragma unroll
        for (int c = 0; c < 8; ++c) {
            const float4 x = rw[c * 32 + lane];   // global_load_b128
            const float4 y = rb[c * 32 + lane];
            accw[c].x += x.x; accw[c].y += x.y; accw[c].z += x.z; accw[c].w += x.w;
            accb[c].x += y.x; accb[c].y += y.y; accb[c].z += y.z; accb[c].w += y.w;
        }
    }

    // ---- activation, stm-ordered output weights, stage prod to LDS --------
    const bool white = stm[b] != 0;
    const float4* owW = (const float4*)(ow + (white ? 0 : HIDDEN));
    const float4* owB = (const float4*)(ow + (white ? HIDDEN : 0));

#pragma unroll
    for (int c = 0; c < 8; ++c) {
        const float4 wv = owW[c * 32 + lane];
        const float4 bv = owB[c * 32 + lane];
        v4h pw, pb;
        float t;
        t = fminf(fmaxf(accw[c].x, 0.f), 1.f); pw[0] = (_Float16)(t * t * wv.x);
        t = fminf(fmaxf(accw[c].y, 0.f), 1.f); pw[1] = (_Float16)(t * t * wv.y);
        t = fminf(fmaxf(accw[c].z, 0.f), 1.f); pw[2] = (_Float16)(t * t * wv.z);
        t = fminf(fmaxf(accw[c].w, 0.f), 1.f); pw[3] = (_Float16)(t * t * wv.w);
        t = fminf(fmaxf(accb[c].x, 0.f), 1.f); pb[0] = (_Float16)(t * t * bv.x);
        t = fminf(fmaxf(accb[c].y, 0.f), 1.f); pb[1] = (_Float16)(t * t * bv.y);
        t = fminf(fmaxf(accb[c].z, 0.f), 1.f); pb[2] = (_Float16)(t * t * bv.z);
        t = fminf(fmaxf(accb[c].w, 0.f), 1.f); pb[3] = (_Float16)(t * t * bv.w);
        // ds_store_b64, conflict-free (32 lanes x 8B = 64 banks)
        *(v4h*)&sprod[wave][c * 128 + lane * 4]          = pw;
        *(v4h*)&sprod[wave][HIDDEN + c * 128 + lane * 4] = pb;
    }
    __syncthreads();

    // ---- output reduction via WMMA: D = A[16x32] x ones[32x16] (+C) -------
    // Every D column equals the per-batch row-sum; wave w reduces K-chunks
    // 4w..4w+3; lane 0 / lane 16 carry the 16 results (M = vgpr + 8*(lane/16)).
    v16h bones;
#pragma unroll
    for (int i = 0; i < 16; ++i) bones[i] = (_Float16)1.0f;

    v8f acc = {0.f, 0.f, 0.f, 0.f, 0.f, 0.f, 0.f, 0.f};
    const int row  = lane & 15;           // A-matrix M for this lane
    const int kOff = (lane >> 4) * 8;     // K group: lanes 16-31 hold K+8

#pragma unroll
    for (int t = 0; t < 4; ++t) {
        const int kc  = wave * 4 + t;     // 64 K-chunks over 16 waves
        const int col = kc * 32 + kOff;
        // A layout (16-bit A 16x32): VGPR0..3 = K[g..g+7], VGPR4..7 = K[g+16..g+23]
        const v8h lo = *(const v8h*)&sprod[row][col];        // ds_load_b128
        const v8h hi = *(const v8h*)&sprod[row][col + 16];   // ds_load_b128
        v16h aM;
#pragma unroll
        for (int i = 0; i < 8; ++i) { aM[i] = lo[i]; aM[8 + i] = hi[i]; }
        acc = __builtin_amdgcn_wmma_f32_16x16x32_f16(
            /*neg_a=*/false, aM, /*neg_b=*/false, bones,
            /*c_mod=*/(short)0, acc, /*reuse_a=*/false, /*reuse_b=*/false);
    }

    if ((lane & 15) == 0) {
        const int mb = (lane >> 4) * 8;   // lane 0 -> M 0..7, lane 16 -> M 8..15
#pragma unroll
        for (int r = 0; r < 8; ++r) atomicAdd(&sOut[mb + r], acc[r]);  // ds_add_f32
    }
    __syncthreads();

    if (tid < BPB) out[blockIdx.x * BPB + tid] = sOut[tid] + ob[0];
}

extern "C" void kernel_launch(void* const* d_in, const int* in_sizes, int n_in,
                              void* d_out, int out_size, void* d_ws, size_t ws_size,
                              hipStream_t stream) {
    const int*   fw  = (const int*)d_in[0];
    const int*   fb  = (const int*)d_in[1];
    const unsigned char* stm = (const unsigned char*)d_in[2];  // jnp.bool_ = 1 byte
    const float* Ww  = (const float*)d_in[3];
    const float* bw  = (const float*)d_in[4];
    const float* Wb  = (const float*)d_in[5];
    const float* bb  = (const float*)d_in[6];
    const float* ow  = (const float*)d_in[7];
    const float* ob  = (const float*)d_in[8];
    float*       out = (float*)d_out;

    const int batch = in_sizes[0] / MAXA;   // 16384
    nnue_ft_kernel<<<dim3(batch / BPB), dim3(512), 0, stream>>>(
        fw, fb, stm, Ww, bw, Wb, bb, ow, ob, out);
}
